// SSEN_80049600462925
// MI455X (gfx1250) — matrix operand order repaired and checked by reading, not compiled, not run on hardware
//
#include <hip/hip_runtime.h>

// ---------------------------------------------------------------------------
// Problem constants (reference): B=4, C=64, H=W=128, 3x3 kernels.
// ---------------------------------------------------------------------------
#define BB   4
#define CC   64
#define HH   128
#define WW   128
#define CI2  128      // 2*C, input channels of est/offset convs
#define KOFF 18       // offset channels

typedef __attribute__((ext_vector_type(16))) __bf16 v16bf;
typedef __attribute__((ext_vector_type(8)))  __bf16 v8bf;
typedef __attribute__((ext_vector_type(8)))  float  v8f;
typedef int v4i __attribute__((vector_size(16)));   // matches builtin signature

#if __has_builtin(__builtin_amdgcn_global_load_async_to_lds_b128)
#define HAVE_ASYNC_LDS 1
#else
#define HAVE_ASYNC_LDS 0
#endif

// 16-byte global->LDS copy; async (ASYNCcnt) when the toolchain exposes it.
static __device__ __forceinline__ void cp16_g2l(const __bf16* g, __bf16* l) {
#if HAVE_ASYNC_LDS
  __builtin_amdgcn_global_load_async_to_lds_b128(
      (__attribute__((address_space(1))) v4i*)g,
      (__attribute__((address_space(3))) v4i*)l, 0, 0);
#else
  *(v8bf*)l = *(const v8bf*)g;
#endif
}

static __device__ __forceinline__ void wait_async_then_barrier() {
#if HAVE_ASYNC_LDS
#if __has_builtin(__builtin_amdgcn_s_wait_asynccnt)
  __builtin_amdgcn_s_wait_asynccnt(0);
#else
  asm volatile("s_wait_asynccnt 0" ::: "memory");
#endif
#endif
  __syncthreads();
}

// A-matrix 16x32 bf16 fragment (ISA 7.12.2): lane<16 holds K {0..7, 16..23},
// lane>=16 holds K {8..15, 24..31}. Caller passes p = row + kb + hi*8.
static __device__ __forceinline__ v16bf ld16_gap(const __bf16* p) {
  union { v16bf v; v8bf h[2]; } u;
  u.h[0] = *(const v8bf*)(p);
  u.h[1] = *(const v8bf*)(p + 16);
  return u.v;
}

// A/B bf16 fragment, contiguous k order used here for both: lane<16 holds K 0..15,
// lane>=16 holds K 16..31. Caller passes p = col_base + kb + hi*16.
static __device__ __forceinline__ v16bf ld16_seq(const __bf16* p) {
  union { v16bf v; v8bf h[2]; } u;
  u.h[0] = *(const v8bf*)(p);
  u.h[1] = *(const v8bf*)(p + 8);
  return u.v;
}

// ---------------------------------------------------------------------------
// NCHW f32 -> NHWC bf16 (LDS-tiled transpose, one (b,y) row per block).
// ---------------------------------------------------------------------------
__global__ __launch_bounds__(256) void nchw2nhwc_k(
    const float* __restrict__ in, __bf16* __restrict__ out) {
  __shared__ __align__(16) __bf16 t[WW * CC];
  int b = blockIdx.x / HH, y = blockIdx.x % HH;
  const float* src = in + ((size_t)b * CC) * (HH * WW) + (size_t)y * WW;
  for (int idx = threadIdx.x; idx < WW * CC; idx += 256) {
    int c = idx >> 7, x = idx & (WW - 1);          // coalesced over x
    t[x * CC + c] = (__bf16)src[(size_t)c * HH * WW + x];
  }
  __syncthreads();
  __bf16* dst = out + (((size_t)b * HH + y) * WW) * CC;
  for (int idx = threadIdx.x; idx < (WW * CC) / 8; idx += 256)
    ((v8bf*)dst)[idx] = ((const v8bf*)t)[idx];     // coalesced 16B stores
}

// ---------------------------------------------------------------------------
// Weight repack: w[Co][Ci][3][3] f32 -> wp[CoPad][9*Ci] bf16, K = tap*Ci+ci
// ---------------------------------------------------------------------------
__global__ void pack_w_k(const float* __restrict__ w, __bf16* __restrict__ wp,
                         int Co, int CoPad, int Ci) {
  int idx = blockIdx.x * blockDim.x + threadIdx.x;
  int total = CoPad * 9 * Ci;
  if (idx >= total) return;
  int co  = idx / (9 * Ci);
  int rem = idx - co * (9 * Ci);
  int tap = rem / Ci;
  int ci  = rem - tap * Ci;
  float v = 0.f;
  if (co < Co) v = w[((size_t)(co * Ci + ci)) * 9 + tap];
  wp[(size_t)co * (9 * Ci) + (size_t)tap * Ci + ci] = (__bf16)v;
}

// ---------------------------------------------------------------------------
// 3x3 conv, stride 1, pad 1, Ci=128 from NHWC bf16 inputs (in0 has nc0
// channels, in1 the rest — the concat). Implicit GEMM, bf16 WMMA.
// One block = (b, y, 16-px x tile); one wave per 16-Co tile.
// Output: NHWC bf16 (out_bf) or NHWC f32 (out_f32).
// ---------------------------------------------------------------------------
__global__ __launch_bounds__(256) void conv3x3_wmma_k(
    const __bf16* __restrict__ in0, const __bf16* __restrict__ in1, int nc0,
    const __bf16* __restrict__ wp,        // [co_tiles*16][9*CI2]
    __bf16* __restrict__ out_bf, float* __restrict__ out_f32,
    int co_tiles, int co_valid, int do_relu) {
  __shared__ __align__(16) __bf16 s_in[3 * 18 * CI2];  // [row][xl][ci]

  int bid = blockIdx.x;
  int xt  = bid & 7;
  int y   = (bid >> 3) & (HH - 1);
  int b   = bid >> 10;
  int x0  = xt << 4;
  int tid = threadIdx.x;
  int nc1 = CI2 - nc0;

  // Stage halo tile: 54 pixels x 16 chunks of 8 channels (16B each).
  for (int idx = tid; idx < 54 * 16; idx += 256) {
    int pix = idx >> 4;
    int ci0 = (idx & 15) << 3;
    int r = pix / 18, xl = pix - r * 18;
    int gy = y + r - 1, gx = x0 + xl - 1;
    __bf16* ldst = &s_in[pix * CI2 + ci0];
    if (gy >= 0 && gy < HH && gx >= 0 && gx < WW) {
      const __bf16* g;
      if (ci0 < nc0) g = in0 + (((size_t)b * HH + gy) * WW + gx) * nc0 + ci0;
      else           g = in1 + (((size_t)b * HH + gy) * WW + gx) * nc1 + (ci0 - nc0);
      cp16_g2l(g, ldst);
    } else {
      v8bf z;
      for (int t = 0; t < 8; ++t) z[t] = (__bf16)0.f;
      *(v8bf*)ldst = z;
    }
  }
  wait_async_then_barrier();

  int wave = tid >> 5;
  int lane = tid & 31;
  if (wave >= co_tiles) return;
  int nl = lane & 15;
  int hi = lane >> 4;
  int co0 = wave << 4;

  v8f acc;
  for (int i = 0; i < 8; ++i) acc[i] = 0.f;

  const __bf16* wrow = wp + (size_t)(co0 + nl) * (9 * CI2);
  for (int tap = 0; tap < 9; ++tap) {
    int dy  = tap / 3;
    int dxx = tap - dy * 3;
    const __bf16* bcol = &s_in[(dy * 18 + nl + dxx) * CI2];
    for (int cc = 0; cc < CI2; cc += 32) {
      int kb = tap * CI2 + cc;
      v16bf a  = ld16_seq(wrow + kb + hi * 16);
      v16bf bm = ld16_seq(bcol + cc + hi * 16);
      acc = __builtin_amdgcn_wmma_f32_16x16x32_bf16(
          false, a, false, bm, (short)0, acc, false, false);
    }
  }

  // C/D: VGPR j = channel co0 + j + hi*8  -> 8 consecutive channels per lane.
  size_t pixbase = ((size_t)b * HH + y) * WW + x0 + nl;
  if (out_bf) {
    v8bf r;
    for (int j = 0; j < 8; ++j) {
      float v = acc[j];
      if (do_relu) v = v > 0.f ? v : 0.f;
      r[j] = (__bf16)v;
    }
    *(v8bf*)&out_bf[pixbase * co_valid + co0 + hi * 8] = r;   // one 16B store
  } else {
    for (int j = 0; j < 8; ++j) {
      int co = co0 + hi * 8 + j;
      if (co < co_valid) {
        float v = acc[j];
        if (do_relu) v = v > 0.f ? v : 0.f;
        out_f32[pixbase * co_valid + co] = v;
      }
    }
  }
}

// ---------------------------------------------------------------------------
// Deformable 3x3 conv. hr: NHWC bf16; off: NHWC f32. Build bilinear samples
// s[16 px][K = tap*64 + c] in LDS (channel-contiguous gathers), then
// 4 waves x 18 bf16 WMMAs. Output NHWC bf16 (intermediate) or NCHW f32.
// ---------------------------------------------------------------------------
__global__ __launch_bounds__(128) void deform_wmma_k(
    const __bf16* __restrict__ hr, const float* __restrict__ off,
    const __bf16* __restrict__ wp,        // [64][9*64]
    __bf16* __restrict__ out_bf, float* __restrict__ out_f32) {
  __shared__ __align__(16) __bf16 s_s[16 * 9 * CC];  // [p][tap*64+c]

  int bid = blockIdx.x;
  int xt  = bid & 7;
  int y   = (bid >> 3) & (HH - 1);
  int b   = bid >> 10;
  int x0  = xt << 4;
  int tid = threadIdx.x;

  const __bf16* hrb = hr + ((size_t)b * HH * WW) * CC;

  for (int pt = tid; pt < 144; pt += 128) {
    int p   = pt / 9;
    int tap = pt - p * 9;
    int xx  = x0 + p;
    size_t obase = (((size_t)b * HH + y) * WW + xx) * KOFF;
    float ody = off[obase + 2 * tap + 0];
    float odx = off[obase + 2 * tap + 1];
    float py = (float)y  + (float)(tap / 3 - 1) + ody;
    float px = (float)xx + (float)(tap % 3 - 1) + odx;
    float y0f = floorf(py), x0f = floorf(px);
    float wy1 = py - y0f,   wx1 = px - x0f;
    float wy0 = 1.f - wy1,  wx0 = 1.f - wx1;
    float vy0 = (y0f >= 0.f       && y0f <= 127.f) ? 1.f : 0.f;
    float vy1 = (y0f + 1.f >= 0.f && y0f + 1.f <= 127.f) ? 1.f : 0.f;
    float vx0 = (x0f >= 0.f       && x0f <= 127.f) ? 1.f : 0.f;
    float vx1 = (x0f + 1.f >= 0.f && x0f + 1.f <= 127.f) ? 1.f : 0.f;
    float w00 = wy0 * wx0 * vy0 * vx0;
    float w01 = wy0 * wx1 * vy0 * vx1;
    float w10 = wy1 * wx0 * vy1 * vx0;
    float w11 = wy1 * wx1 * vy1 * vx1;
    int yc0 = (int)fminf(fmaxf(y0f,       0.f), 127.f);
    int yc1 = (int)fminf(fmaxf(y0f + 1.f, 0.f), 127.f);
    int xc0 = (int)fminf(fmaxf(x0f,       0.f), 127.f);
    int xc1 = (int)fminf(fmaxf(x0f + 1.f, 0.f), 127.f);
    const __bf16* p00 = hrb + ((size_t)yc0 * WW + xc0) * CC;
    const __bf16* p01 = hrb + ((size_t)yc0 * WW + xc1) * CC;
    const __bf16* p10 = hrb + ((size_t)yc1 * WW + xc0) * CC;
    const __bf16* p11 = hrb + ((size_t)yc1 * WW + xc1) * CC;
    __bf16* dst = &s_s[p * (9 * CC) + tap * CC];
    for (int cc = 0; cc < CC; cc += 8) {   // channel-contiguous 16B loads
      v8bf a00 = *(const v8bf*)(p00 + cc);
      v8bf a01 = *(const v8bf*)(p01 + cc);
      v8bf a10 = *(const v8bf*)(p10 + cc);
      v8bf a11 = *(const v8bf*)(p11 + cc);
      v8bf r;
      for (int t = 0; t < 8; ++t) {
        float v = w00 * (float)a00[t] + w01 * (float)a01[t] +
                  w10 * (float)a10[t] + w11 * (float)a11[t];
        r[t] = (__bf16)v;
      }
      *(v8bf*)(dst + cc) = r;
    }
  }
  __syncthreads();

  int wave = tid >> 5;
  int lane = tid & 31;
  int nl = lane & 15;
  int hi = lane >> 4;
  int co0 = wave << 4;

  v8f acc;
  for (int i = 0; i < 8; ++i) acc[i] = 0.f;

  const __bf16* wrow = wp + (size_t)(co0 + nl) * (9 * CC);
  const __bf16* bcol = &s_s[nl * (9 * CC)];
  for (int kb = 0; kb < 9 * CC; kb += 32) {
    v16bf a  = ld16_seq(wrow + kb + hi * 16);
    v16bf bm = ld16_seq(bcol + kb + hi * 16);
    acc = __builtin_amdgcn_wmma_f32_16x16x32_bf16(
        false, a, false, bm, (short)0, acc, false, false);
  }

  if (out_bf) {
    v8bf r;
    for (int j = 0; j < 8; ++j) r[j] = (__bf16)acc[j];
    *(v8bf*)&out_bf[(((size_t)b * HH + y) * WW + x0 + nl) * CC + co0 + hi * 8] = r;
  } else {
    for (int j = 0; j < 8; ++j) {
      int co = co0 + hi * 8 + j;
      out_f32[(((size_t)b * CC + co) * HH + y) * WW + x0 + nl] = acc[j];
    }
  }
}

// ---------------------------------------------------------------------------
// Orchestration: convert lr/hr0 to NHWC bf16 once, then 3 blocks of
// (pack weights, est conv, offset conv, deformable conv). Graph-capture safe.
// ---------------------------------------------------------------------------
extern "C" void kernel_launch(void* const* d_in, const int* in_sizes, int n_in,
                              void* d_out, int out_size, void* d_ws, size_t ws_size,
                              hipStream_t stream) {
  (void)in_sizes; (void)n_in; (void)out_size; (void)ws_size;
  const float* lr  = (const float*)d_in[0];
  const float* hr0 = (const float*)d_in[1];
  float* out = (float*)d_out;

  char* ws = (char*)d_ws;
  size_t o = 0;
  __bf16* lrbf  = (__bf16*)(ws + o); o += (size_t)BB * HH * WW * CC  * 2;  // 8.4 MB
  __bf16* hrbf0 = (__bf16*)(ws + o); o += (size_t)BB * HH * WW * CC  * 2;
  __bf16* hrbfA = (__bf16*)(ws + o); o += (size_t)BB * HH * WW * CC  * 2;
  __bf16* hrbfB = (__bf16*)(ws + o); o += (size_t)BB * HH * WW * CC  * 2;
  __bf16* estbf = (__bf16*)(ws + o); o += (size_t)BB * HH * WW * CI2 * 2;  // 16.8 MB
  float*  offf  = (float*) (ws + o); o += (size_t)BB * HH * WW * KOFF * 4; //  4.7 MB
  __bf16* wpe   = (__bf16*)(ws + o); o += (size_t)128 * (9 * CI2) * 2;
  __bf16* wpo   = (__bf16*)(ws + o); o += (size_t)32  * (9 * CI2) * 2;
  __bf16* wpd   = (__bf16*)(ws + o); o += (size_t)64  * (9 * CC)  * 2;

  nchw2nhwc_k<<<BB * HH, 256, 0, stream>>>(lr,  lrbf);
  nchw2nhwc_k<<<BB * HH, 256, 0, stream>>>(hr0, hrbf0);

  const int grid = BB * HH * (WW / 16);   // 4096 tiles
  const __bf16* hr_prev = hrbf0;
  __bf16* hr_bf_next[3] = { hrbfA, hrbfB, nullptr };

  for (int it = 0; it < 3; ++it) {
    const float* ew = (const float*)d_in[2 + 3 * it];
    const float* ow = (const float*)d_in[3 + 3 * it];
    const float* dw = (const float*)d_in[4 + 3 * it];

    pack_w_k<<<(128 * 9 * CI2 + 255) / 256, 256, 0, stream>>>(ew, wpe, 128, 128, CI2);
    pack_w_k<<<(32  * 9 * CI2 + 255) / 256, 256, 0, stream>>>(ow, wpo, KOFF, 32, CI2);
    pack_w_k<<<(64  * 9 * CC  + 255) / 256, 256, 0, stream>>>(dw, wpd, CC, CC, CC);

    // est = relu(conv3x3(cat(lr, hr_prev))) -> NHWC bf16, Co=128
    conv3x3_wmma_k<<<grid, 256, 0, stream>>>(lrbf, hr_prev, CC, wpe,
                                             estbf, nullptr, 8, CI2, 1);
    // offset = conv3x3(est) -> NHWC f32, Co=18 (padded to 32)
    conv3x3_wmma_k<<<grid, 256, 0, stream>>>(estbf, estbf, CI2, wpo,
                                             nullptr, offf, 2, KOFF, 0);
    // hr_next = deform_conv3x3(hr_prev, offset)
    deform_wmma_k<<<grid, 128, 0, stream>>>(hr_prev, offf, wpd,
                                            hr_bf_next[it],
                                            it == 2 ? out : nullptr);
    hr_prev = hr_bf_next[it];
  }
}
